// BahdanauAttention_5454608466757
// MI455X (gfx1250) — matrix operand order repaired and checked by reading, not compiled, not run on hardware
//
#include <hip/hip_runtime.h>
#include <cmath>

#define HH 1024
#define DD 2048   // 2H
#define BB 64
#define SS 2048
#define LDA 2056  // padded bf16 row stride: 4112B -> lane bank = 4*nh (conflict-free)

typedef __attribute__((ext_vector_type(16))) __bf16 v16bf;
typedef __attribute__((ext_vector_type(8)))  __bf16 v8bf;
typedef __attribute__((ext_vector_type(4)))  __bf16 v4bf;
typedef __attribute__((ext_vector_type(8)))  float  v8f;

static __device__ __forceinline__ v16bf cat8(v8bf lo, v8bf hi) {
  v16bf r;
#pragma unroll
  for (int i = 0; i < 8; ++i) { r[i] = lo[i]; r[i + 8] = hi[i]; }
  return r;
}

// ---------- k1: convert W_h (H x 2H f32) -> bf16 workspace ----------
__global__ void k_cvt_wh(const float* __restrict__ wh, __bf16* __restrict__ whb) {
  int i8 = (blockIdx.x * 256 + threadIdx.x) * 8;
  float4 a = ((const float4*)wh)[i8 / 4];
  float4 b = ((const float4*)wh)[i8 / 4 + 1];
  v8bf o;
  o[0] = (__bf16)a.x; o[1] = (__bf16)a.y; o[2] = (__bf16)a.z; o[3] = (__bf16)a.w;
  o[4] = (__bf16)b.x; o[5] = (__bf16)b.y; o[6] = (__bf16)b.z; o[7] = (__bf16)b.w;
  *(v8bf*)(whb + i8) = o;
}

// ---------- k2: Wsb[b][h] = s_t[b] . W_s[h] + b_s[h] + b_h[h]  (exact f32) ----------
__global__ void k_wsb(const float* __restrict__ s_t, const float* __restrict__ W_s,
                      const float* __restrict__ b_s, const float* __restrict__ b_h,
                      float* __restrict__ Wsb) {
  __shared__ float4 sl[DD / 4];
  const int b = blockIdx.x, tid = threadIdx.x;
  const float4* sr = (const float4*)(s_t + (size_t)b * DD);
#pragma unroll
  for (int i = 0; i < 2; ++i) sl[tid + 256 * i] = sr[tid + 256 * i];
  __syncthreads();
#pragma unroll
  for (int j = 0; j < HH / 256; ++j) {
    int h = tid + 256 * j;
    const float4* wr = (const float4*)(W_s + (size_t)h * DD);
    float acc = 0.f;
    for (int k = 0; k < DD / 4; ++k) {
      float4 sv = sl[k], wv = wr[k];
      acc += sv.x * wv.x + sv.y * wv.y + sv.z * wv.z + sv.w * wv.w;
    }
    Wsb[b * HH + h] = acc + b_s[h] + b_h[h];
  }
}

// ---------- k3: fused score GEMM + tanh + v_a reduction -> E (bf16 WMMA, f32 acc) ----------
__global__ void __launch_bounds__(256, 1) k_scores(
    const float* __restrict__ h_i, const __bf16* __restrict__ whb,
    const float* __restrict__ Wsb, const float* __restrict__ coverage,
    const float* __restrict__ W_c, const float* __restrict__ v_a,
    const float* __restrict__ b_a, float* __restrict__ Eout) {
  extern __shared__ char smem[];
  __bf16* At  = (__bf16*)smem;                       // 64 x LDA bf16
  float* eacc = (float*)(smem + (size_t)64 * LDA * 2);  // 64
  float* covs = eacc + 64;                              // 64
  const int tid = threadIdx.x;
  const int b  = blockIdx.y;
  const int s0 = blockIdx.x * 64;

  if (tid < 64) { eacc[tid] = 0.f; covs[tid] = coverage[b * SS + s0 + tid]; }

  // stage 64 rows x 2048 of h_i -> bf16 LDS (f32x4 loads, ds_store_b64)
  {
    const float4* src = (const float4*)(h_i + (size_t)(b * SS + s0) * DD);
    for (int i = 0; i < 128; ++i) {
      int f4 = tid + 256 * i;           // 64 rows * 512 float4
      int row = f4 >> 9, c4 = f4 & 511;
      float4 v = src[(size_t)row * 512 + c4];
      v4bf o;
      o[0] = (__bf16)v.x; o[1] = (__bf16)v.y; o[2] = (__bf16)v.z; o[3] = (__bf16)v.w;
      *(v4bf*)(At + row * LDA + c4 * 4) = o;
    }
  }
  __syncthreads();

  const int lane = tid & 31;
  const int wave = tid >> 5;
  const int nh   = lane & 15;   // N column within tile / M row for A
  const int half = lane >> 4;   // lane half selects K sub-range

  for (int j = wave; j < 16; j += 8) {      // 16 h-slices of 64, 2 per wave
    const int h_base = j * 64;
    v8f z = {0.f, 0.f, 0.f, 0.f, 0.f, 0.f, 0.f, 0.f};
    v8f acc[4][4];
#pragma unroll
    for (int s = 0; s < 4; ++s)
#pragma unroll
      for (int t = 0; t < 4; ++t) acc[s][t] = z;

    const __bf16* wb0 = whb + (size_t)(h_base + nh) * DD + 16 * half;
    for (int kc = 0; kc < DD; kc += 32) {
      v16bf bf[4];
#pragma unroll
      for (int t = 0; t < 4; ++t) {                      // B frag: 16 contiguous K
        const __bf16* p = wb0 + (size_t)t * 16 * DD + kc;
        bf[t] = cat8(*(const v8bf*)p, *(const v8bf*)(p + 8));
      }
#pragma unroll
      for (int s = 0; s < 4; ++s) {                      // A frag: 16 contiguous K per half
        const __bf16* ap = At + (s * 16 + nh) * LDA + kc + 16 * half;
        v16bf a = cat8(*(const v8bf*)ap, *(const v8bf*)(ap + 8));
#pragma unroll
        for (int t = 0; t < 4; ++t)
          acc[s][t] = __builtin_amdgcn_wmma_f32_16x16x32_bf16(
              false, a, false, bf[t], (short)0, acc[s][t], false, false);
      }
    }

    // epilogue: E[M] += sum_N tanh(acc + Wsb + cov*W_c) * v_a
#pragma unroll
    for (int t = 0; t < 4; ++t) {
      int h = h_base + t * 16 + nh;
      float wsb = Wsb[b * HH + h];
      float wc  = W_c[h];
      float va  = v_a[h];
#pragma unroll
      for (int s = 0; s < 4; ++s) {
#pragma unroll
        for (int r = 0; r < 8; ++r) {
          int M = s * 16 + r + 8 * half;   // D-layout: VGPR r, lane-half -> M
          float val = acc[s][t][r] + wsb + covs[M] * wc;
          float e = tanhf(val) * va;
          e += __shfl_xor(e, 1, 32);
          e += __shfl_xor(e, 2, 32);
          e += __shfl_xor(e, 4, 32);
          e += __shfl_xor(e, 8, 32);       // reduce 16 N-lanes within half
          if (nh == 0) atomicAdd(&eacc[M], e);
        }
      }
    }
  }
  __syncthreads();
  if (tid < 64) Eout[b * SS + s0 + tid] = eacc[tid] + b_a[0];
}

// ---------- k4: softmax over S (in place on E->A) + new_cov ----------
__global__ void k_softmax(const float* __restrict__ coverage, float* __restrict__ EA,
                          float* __restrict__ outCov) {
  __shared__ float red[256];
  const int b = blockIdx.x, tid = threadIdx.x;
  float v[8];
  float m = -1e30f;
#pragma unroll
  for (int i = 0; i < 8; ++i) { v[i] = EA[b * SS + tid + 256 * i]; m = fmaxf(m, v[i]); }
  red[tid] = m; __syncthreads();
  for (int w = 128; w > 0; w >>= 1) {
    if (tid < w) red[tid] = fmaxf(red[tid], red[tid + w]);
    __syncthreads();
  }
  m = red[0]; __syncthreads();
  float sum = 0.f;
#pragma unroll
  for (int i = 0; i < 8; ++i) { v[i] = expf(v[i] - m); sum += v[i]; }
  red[tid] = sum; __syncthreads();
  for (int w = 128; w > 0; w >>= 1) {
    if (tid < w) red[tid] += red[tid + w];
    __syncthreads();
  }
  float inv = 1.f / red[0];
#pragma unroll
  for (int i = 0; i < 8; ++i) {
    int idx = b * SS + tid + 256 * i;
    float a = v[i] * inv;
    EA[idx] = a;
    outCov[idx] = coverage[idx] + a;
  }
}

// ---------- k5: C[b][d] = sum_s A[b][s] * h_i[b][s][d]  (memory-bound, exact f32) ----------
__global__ void k_context(const float* __restrict__ h_i, const float* __restrict__ A,
                          float* __restrict__ C) {
  const int b = blockIdx.y;
  const int d = blockIdx.x * 256 + threadIdx.x;
  const float* hp = h_i + (size_t)b * SS * DD + d;
  const float* ap = A + b * SS;
  float acc = 0.f;
  for (int s = 0; s < SS; s += 4) {
    acc += ap[s + 0] * hp[(size_t)(s + 0) * DD];
    acc += ap[s + 1] * hp[(size_t)(s + 1) * DD];
    acc += ap[s + 2] * hp[(size_t)(s + 2) * DD];
    acc += ap[s + 3] * hp[(size_t)(s + 3) * DD];
  }
  C[b * DD + d] = acc;
}

extern "C" void kernel_launch(void* const* d_in, const int* in_sizes, int n_in,
                              void* d_out, int out_size, void* d_ws, size_t ws_size,
                              hipStream_t stream) {
  const float* s_t = (const float*)d_in[0];
  const float* h_i = (const float*)d_in[1];
  const float* cov = (const float*)d_in[2];
  const float* W_h = (const float*)d_in[3];
  const float* b_h = (const float*)d_in[4];
  const float* W_s = (const float*)d_in[5];
  const float* b_s = (const float*)d_in[6];
  const float* W_c = (const float*)d_in[7];
  const float* v_a = (const float*)d_in[8];
  const float* b_a = (const float*)d_in[9];

  float* outC   = (float*)d_out;            // (B, 2H)
  float* outA   = outC + BB * DD;           // (B, S) — also holds E pre-softmax
  float* outCov = outA + BB * SS;           // (B, S)

  __bf16* whb = (__bf16*)d_ws;                                  // 4 MB
  float*  Wsb = (float*)((char*)d_ws + (size_t)HH * DD * 2);    // 256 KB

  k_cvt_wh<<<1024, 256, 0, stream>>>(W_h, whb);
  k_wsb<<<BB, 256, 0, stream>>>(s_t, W_s, b_s, b_h, Wsb);

  size_t smem = (size_t)64 * LDA * 2 + 128 * sizeof(float);     // ~258 KB (<= 320 KB/WGP)
  hipFuncSetAttribute((const void*)k_scores,
                      hipFuncAttributeMaxDynamicSharedMemorySize, (int)smem);
  k_scores<<<dim3(32, BB), 256, smem, stream>>>(h_i, whb, Wsb, cov, W_c, v_a, b_a, outA);

  k_softmax<<<BB, 256, 0, stream>>>(cov, outA, outCov);
  k_context<<<dim3(DD / 256, BB), 256, 0, stream>>>(h_i, outA, outC);
}
